// RNN_5136780886338
// MI455X (gfx1250) — matrix operand order repaired and checked by reading, not compiled, not run on hardware
//
#include <hip/hip_runtime.h>

// LSTM char-RNN on gfx1250 (MI455X), wave32 + v_wmma_f32_16x16x32_bf16.
//
//  * Pre-pass builds W_allT (bf16, [1024][384]) = [W_h^T | W_x^T] in d_ws.
//  * Main kernel: 32 WGs x 512 threads (16 waves). Each WG owns 16 batch rows
//    for all 1024 timesteps. A matrix = [h | onehot(token)] (16x384 bf16) in
//    LDS; cell state c and all 4 gate accumulators live in registers.
//  * W_allT stays resident in the 192MB L2. B fragments are re-read each
//    timestep through an LICM-opaque, address_space(1) base (lowers to
//    global_load_b128, LOADcnt-only), with a distance-2 rotating prefetch
//    pipeline to hide L2 latency within a single wave.

#define SEQ_LEN 1024
#define UNITS_  256
#define NCH_    128
#define KTOT    384          // 256 recurrent + 128 one-hot K columns
#define HSTR    392          // padded LDS row stride (shorts); 392*2=784B, 16B aligned
#define BT      16           // batch rows per workgroup
#define WGT     512          // threads per workgroup (16 waves)

#define GLOBAL_AS __attribute__((address_space(1)))

typedef __attribute__((ext_vector_type(16))) __bf16 v16bf;
typedef __attribute__((ext_vector_type(8)))  float  v8f;
typedef __attribute__((ext_vector_type(4)))  unsigned v4u;   // POD 128-bit carrier

union ABFrag { v4u u[2]; v16bf v; };

__device__ __forceinline__ unsigned short f2bf(float f) {
  union { float f; unsigned u; } v; v.f = f;
  unsigned r = v.u + 0x7FFFu + ((v.u >> 16) & 1u);   // round-to-nearest-even
  return (unsigned short)(r >> 16);
}
__device__ __forceinline__ float bf2f(unsigned short h) {
  union { unsigned u; float f; } v; v.u = ((unsigned)h) << 16;
  return v.f;
}
__device__ __forceinline__ float sigm(float x) { return 1.0f / (1.0f + __expf(-x)); }

// ---------------------------------------------------------------------------
// Pre-pass: W_allT[n][k] (bf16) = k<256 ? W_h[k][n] : W_x[k-256][n]
// ---------------------------------------------------------------------------
__global__ void build_wallt(const float* __restrict__ Wx,
                            const float* __restrict__ Wh,
                            unsigned short* __restrict__ WallT) {
  int n = blockIdx.x;      // 0..1023 (gate column)
  int k = threadIdx.x;     // 0..383
  float v = (k < 256) ? Wh[(size_t)k * 1024 + n] : Wx[(size_t)(k - 256) * 1024 + n];
  WallT[(size_t)n * KTOT + k] = f2bf(v);
}

// ---------------------------------------------------------------------------
// Main fused LSTM kernel
// ---------------------------------------------------------------------------
__global__ __launch_bounds__(WGT)
void lstm_wmma_kernel(const int* __restrict__ tokens,
                      const float* __restrict__ bias,
                      const float* __restrict__ Wdense,
                      const float* __restrict__ bdense,
                      const unsigned short* __restrict__ WallT,
                      float* __restrict__ out) {
  __shared__ unsigned short hbuf[BT * HSTR];   // [16 rows][h(256) | onehot(128) | pad]

  const int tid     = threadIdx.x;
  const int lane    = tid & 31;
  const int wave    = tid >> 5;        // 0..15
  const int wgBase  = blockIdx.x * BT; // first batch row of this WG
  const int halfSel = lane >> 4;       // 0: lanes 0-15, 1: lanes 16-31
  const int nloc    = lane & 15;
  const int u       = wave * 16 + nloc;   // unit column owned by this lane

  // zero LDS (h state starts at 0; one-hot region cleared)
  for (int i = tid; i < BT * HSTR; i += WGT) hbuf[i] = 0;

  // biases for the 4 gates at unit u (time-invariant)
  const float b_i = bias[u];
  const float b_j = bias[256 + u];
  const float b_f = bias[512 + u];
  const float b_o = bias[768 + u];

  // Per-lane byte offsets of the 4 gate B-tiles (time-invariant).
  // B layout (32x16 bf16): lane holds 16 contiguous K values; lanes 0-15 get
  // K=ks*32..+15, lanes 16-31 get K=ks*32+16..+31, column = gate tile col + nloc.
  unsigned boff[4];
#pragma unroll
  for (int g = 0; g < 4; ++g)
    boff[g] = (unsigned)(((g * 256 + u) * KTOT + (halfSel << 4)) * sizeof(unsigned short));

  // A layout (16x32 bf16): lanes 0-15: M=lane, K = k0..+15;
  // lanes 16-31: M=lane-16, K = k0+16..+31.
  const int aOff = nloc * HSTR + (halfSel << 4);

  v8f creg = {};   // cell state: element r,lane -> (M = r+8*halfSel, unit u)

  // one-hot writer: 32 threads per batch row, 4 chars each
  const int ohRow = tid >> 5;
  const int ohCol = (tid & 31) << 2;

  __syncthreads();
  { // one-hot for t = 0
    int tok = tokens[(size_t)(wgBase + ohRow) * SEQ_LEN];
    unsigned short* row = hbuf + ohRow * HSTR + 256;
#pragma unroll
    for (int j = 0; j < 4; ++j)
      row[ohCol + j] = ((ohCol + j) == tok) ? (unsigned short)0x3F80 : (unsigned short)0;
  }

#pragma unroll 1
  for (int t = 0; t < SEQ_LEN; ++t) {
    __syncthreads();   // one-hot(t) + h(t) visible to all waves

    // LICM-opaque W base, kept in SGPRs and cast to the GLOBAL address space
    // so B loads lower to global_load_b128 and cannot be hoisted+spilled.
    uintptr_t wa = (uintptr_t)WallT;
    asm volatile("" : "+s"(wa));
    const GLOBAL_AS char* wp = (const GLOBAL_AS char*)wa;

    v8f acc0 = {}, acc1 = {}, acc2 = {}, acc3 = {};

    // Distance-2 rotating prefetch pipeline for B fragments (3 buffers).
    ABFrag bb[3][4];
#pragma unroll
    for (int s = 0; s < 2; ++s) {       // preload K-steps 0 and 1
#pragma unroll
      for (int g = 0; g < 4; ++g) {
        const GLOBAL_AS v4u* p = (const GLOBAL_AS v4u*)(wp + boff[g]) + s * 4;
        bb[s][g].u[0] = p[0]; bb[s][g].u[1] = p[1];
      }
    }

#pragma unroll
    for (int ks = 0; ks < 12; ++ks) {
      if (ks + 2 < 12) {                // prefetch K-step ks+2
        const int slot = (ks + 2) % 3;
#pragma unroll
        for (int g = 0; g < 4; ++g) {
          const GLOBAL_AS v4u* p = (const GLOBAL_AS v4u*)(wp + boff[g]) + (ks + 2) * 4;
          bb[slot][g].u[0] = p[0]; bb[slot][g].u[1] = p[1];
        }
      }

      ABFrag a;
      const v4u* ap = (const v4u*)(hbuf + aOff + ks * 32);
      a.u[0] = ap[0];       // K = ks*32 + halfSel*16 .. +7
      a.u[1] = ap[1];       // K = ks*32 + halfSel*16 + 8 .. +15

      const int cur = ks % 3;
      acc0 = __builtin_amdgcn_wmma_f32_16x16x32_bf16(false, a.v, false, bb[cur][0].v, (short)0, acc0, false, false);
      acc1 = __builtin_amdgcn_wmma_f32_16x16x32_bf16(false, a.v, false, bb[cur][1].v, (short)0, acc1, false, false);
      acc2 = __builtin_amdgcn_wmma_f32_16x16x32_bf16(false, a.v, false, bb[cur][2].v, (short)0, acc2, false, false);
      acc3 = __builtin_amdgcn_wmma_f32_16x16x32_bf16(false, a.v, false, bb[cur][3].v, (short)0, acc3, false, false);
    }

    __syncthreads();   // all waves done reading hbuf before we overwrite h

    // register-resident LSTM cell update: this wave's 16 units x 16 rows
#pragma unroll
    for (int r = 0; r < 8; ++r) {
      int M = r + (halfSel << 3);
      float gi = acc0[r] + b_i;
      float gj = acc1[r] + b_j;
      float gf = acc2[r] + b_f;
      float go = acc3[r] + b_o;
      float nc = creg[r] * sigm(gf + 1.0f) + sigm(gi) * tanhf(gj);  // forget_bias=1
      float nh = tanhf(nc) * sigm(go);
      creg[r] = nc;
      hbuf[M * HSTR + u] = f2bf(nh);
    }

    if (t + 1 < SEQ_LEN) {   // one-hot for t+1
      int tok = tokens[(size_t)(wgBase + ohRow) * SEQ_LEN + (t + 1)];
      unsigned short* row = hbuf + ohRow * HSTR + 256;
#pragma unroll
      for (int j = 0; j < 4; ++j)
        row[ohCol + j] = ((ohCol + j) == tok) ? (unsigned short)0x3F80 : (unsigned short)0;
    }
  }

  __syncthreads();

  // Final dense: out[m][n] = h[m] . Wdense[:,n] + bdense[n]   (16x128 per WG)
  {
    int m  = tid >> 5;            // batch row in tile
    int nb = (tid & 31) << 2;     // 4 output chars per thread
    float4 o;
    o.x = bdense[nb]; o.y = bdense[nb + 1]; o.z = bdense[nb + 2]; o.w = bdense[nb + 3];
#pragma unroll 4
    for (int k = 0; k < UNITS_; ++k) {
      float hv = bf2f(hbuf[m * HSTR + k]);
      float4 w = *(const float4*)(Wdense + (size_t)k * NCH_ + nb);
      o.x += hv * w.x; o.y += hv * w.y; o.z += hv * w.z; o.w += hv * w.w;
    }
    *(float4*)(out + (size_t)(wgBase + m) * NCH_ + nb) = o;
  }
}

// ---------------------------------------------------------------------------
extern "C" void kernel_launch(void* const* d_in, const int* in_sizes, int n_in,
                              void* d_out, int out_size, void* d_ws, size_t ws_size,
                              hipStream_t stream) {
  (void)in_sizes; (void)n_in; (void)out_size; (void)ws_size;
  const int*   tokens = (const int*)d_in[0];
  const float* W_x    = (const float*)d_in[1];
  const float* W_h    = (const float*)d_in[2];
  const float* b      = (const float*)d_in[3];
  const float* W_d    = (const float*)d_in[4];
  const float* b_d    = (const float*)d_in[5];
  float*       out    = (float*)d_out;
  unsigned short* WallT = (unsigned short*)d_ws;   // 1024*384*2 = 768 KB

  build_wallt<<<dim3(1024), dim3(KTOT), 0, stream>>>(W_x, W_h, WallT);
  lstm_wmma_kernel<<<dim3(512 / BT), dim3(WGT), 0, stream>>>(
      tokens, b, W_d, b_d, WallT, out);
}
